// DenoiseNet_29532195127426
// MI455X (gfx1250) — compile-verified, matrix operand order checked by reading, not run on hardware
//
#include <hip/hip_runtime.h>

typedef __attribute__((ext_vector_type(16))) _Float16 v16h;
typedef __attribute__((ext_vector_type(8)))  _Float16 v8h;
typedef __attribute__((ext_vector_type(8)))  float    v8f;

constexpr int BB  = 4;       // batch
constexpr int NN  = 10000;   // points
constexpr int MM  = 128;     // selected pts
constexpr int KF  = 32;      // frame knn
constexpr int CC  = 4;       // clean nbs
constexpr int FD  = 128;     // feature dim
constexpr int HD  = 128;     // hidden
// 0.5 * (1/sigma) / (B*M*K),  sigma = 0.01
constexpr float LOSS_SCALE = 50.0f / (float)(BB * MM * KF);

// ---------------- WMMA helpers ----------------

__device__ __forceinline__ v8f wmma16(v16h a, v16h b, v8f c) {
    // emits v_wmma_f32_16x16x32_f16
    return __builtin_amdgcn_wmma_f32_16x16x32_f16(false, a, false, b,
                                                  (short)0, c, false, false);
}

// A fragment (16x32 f16): all lanes share one 128-half row (broadcast rows).
// lanes 0-15: K = base+{0..7, 16..23}; lanes 16-31: K = base+{8..15, 24..31}.
__device__ __forceinline__ v16h frag_from_row128(const _Float16* row, int kk, int lane) {
    int k0 = kk * 32 + ((lane >> 4) << 3);
    v8h lo = *(const v8h*)(row + k0);
    v8h hi = *(const v8h*)(row + k0 + 16);
    return __builtin_shufflevector(lo, hi, 0,1,2,3,4,5,6,7,8,9,10,11,12,13,14,15);
}

// A fragment from a [16][128] f16 tile: lane's row = lane%16.
__device__ __forceinline__ v16h frag_a_tile(const _Float16* t, int kk, int lane) {
    return frag_from_row128(t + (lane & 15) * 128, kk, lane);
}

// B fragment from weights pre-swizzled as [(n*4+kk)*2+g][16 halves].
__device__ __forceinline__ v16h frag_b_swz(const _Float16* swz, int c, int kk, int lane) {
    int n = c * 16 + (lane & 15);
    int g = lane >> 4;
    const _Float16* p = swz + ((((n << 2) + kk) * 2 + g) << 4);
    v8h lo = *(const v8h*)(p);
    v8h hi = *(const v8h*)(p + 8);
    return __builtin_shufflevector(lo, hi, 0,1,2,3,4,5,6,7,8,9,10,11,12,13,14,15);
}

// Swizzle a 128x128 (row-major, leading dim ldw, starting at row_off) f32 weight
// matrix into B-fragment order (f16) so fragment loads are contiguous 16B.
__device__ __forceinline__ void load_swz(const float* __restrict__ W, int ldw, int row_off,
                                         _Float16* dst, int tid, int nthr) {
    for (int e = tid; e < 128 * 128; e += nthr) {
        int j = e & 15, g = (e >> 4) & 1, kk = (e >> 5) & 3, n = e >> 7;
        int k = kk * 32 + (g << 3) + (j < 8 ? j : j + 8);
        dst[e] = (_Float16)W[(long)(row_off + k) * ldw + n];
    }
}

__device__ __forceinline__ unsigned long long wave_min_u64(unsigned long long k) {
#pragma unroll
    for (int off = 16; off; off >>= 1) {
        unsigned lo = __shfl_xor((unsigned)k, off, 32);
        unsigned hi = __shfl_xor((unsigned)(k >> 32), off, 32);
        unsigned long long o = ((unsigned long long)hi << 32) | lo;
        k = (o < k) ? o : k;
    }
    return k;
}

// Sorted insert into 4-register ascending list (k0 <= k1 <= k2 <= k3).
__device__ __forceinline__ void ins4(unsigned long long key,
                                     unsigned long long& k0, unsigned long long& k1,
                                     unsigned long long& k2, unsigned long long& k3) {
    if (key < k3) {
        if (key < k2) {
            k3 = k2;
            if (key < k1) {
                k2 = k1;
                if (key < k0) { k1 = k0; k0 = key; } else { k1 = key; }
            } else { k2 = key; }
        } else { k3 = key; }
    }
}

// ---------------- Kernel 1: gather q + feature MLP (only at gathered points) ----------------
// grid 32 x 256; each block: 16 rows of (B*M, F) features.
__global__ __launch_bounds__(256) void k_feat(
    const float* __restrict__ noisy, const int* __restrict__ pidx,
    const float* __restrict__ fW1, const float* __restrict__ fb1,
    const float* __restrict__ fW2, const float* __restrict__ fb2,
    float* __restrict__ qws, _Float16* __restrict__ featF16)
{
    __shared__ float     qs[16][3];
    __shared__ __align__(16) _Float16 hs[16 * 128];
    __shared__ __align__(16) _Float16 w2s[128 * 128];

    int tid = threadIdx.x;
    int r0  = blockIdx.x * 16;

    load_swz(fW2, FD, 0, w2s, tid, 256);

    if (tid < 16) {
        int r = r0 + tid;
        int b = r / MM, m = r % MM;
        int idx = pidx[m];
        const float* p = noisy + ((long)b * NN + idx) * 3;
        float x = p[0], y = p[1], z = p[2];
        qs[tid][0] = x; qs[tid][1] = y; qs[tid][2] = z;
        qws[r * 3 + 0] = x; qws[r * 3 + 1] = y; qws[r * 3 + 2] = z;
    }
    __syncthreads();

    // layer 1: (16,3)@(3,128) + relu  (K=3 -> VALU)
    for (int e = tid; e < 16 * 128; e += 256) {
        int i = e >> 7, n = e & 127;
        float v = fb1[n] + qs[i][0] * fW1[n] + qs[i][1] * fW1[128 + n] + qs[i][2] * fW1[256 + n];
        hs[e] = (_Float16)fmaxf(v, 0.0f);
    }
    __syncthreads();

    // layer 2: (16,128)@(128,128) via WMMA; wave w owns column tile w.
    int wave = tid >> 5, lane = tid & 31;
    int col  = wave * 16 + (lane & 15);
    v8f acc;
    float bias = fb2[col];
#pragma unroll
    for (int v = 0; v < 8; v++) acc[v] = bias;
#pragma unroll
    for (int kk = 0; kk < 4; kk++) {
        v16h a = frag_a_tile(hs, kk, lane);
        v16h b = frag_b_swz(w2s, wave, kk, lane);
        acc = wmma16(a, b, acc);
    }
    int rbase = r0 + ((lane >> 4) << 3);
#pragma unroll
    for (int v = 0; v < 8; v++)
        featF16[(long)(rbase + v) * FD + col] = (_Float16)acc[v];
}

// ---------------- Kernel 2: KNN1 (32 nearest noisy) -> frames, frames_centered ----------------
// One wave per query; 8 waves/block; 16 blocks per batch (128 queries / 8).
// Cloud staged in LDS (SoA). 8 super-passes x (scan keeping per-lane top-4 above
// the exclusion key, then extract the 4 global winners).
__global__ __launch_bounds__(256) void k_knn1(
    const float* __restrict__ noisy, const float* __restrict__ qws,
    float* __restrict__ frames, float* __restrict__ xc)
{
    __shared__ float xs[NN], ys[NN], zs[NN];   // 120 KB SoA cloud

    int tid  = threadIdx.x;
    int b    = blockIdx.x >> 4;                // 16 blocks per batch
    const float* src = noisy + (long)b * NN * 3;
    for (int e = tid; e < NN; e += 256) {
        __builtin_prefetch(src + 3 * e + 768, 0, 0);   // global_prefetch_b8 ahead
        xs[e] = src[3 * e];
        ys[e] = src[3 * e + 1];
        zs[e] = src[3 * e + 2];
    }
    __syncthreads();

    int wave = tid >> 5, lane = tid & 31;
    int w    = blockIdx.x * 8 + wave;          // query index in [0, B*M)
    float qx = qws[w * 3], qy = qws[w * 3 + 1], qz = qws[w * 3 + 2];

    unsigned long long prev = 0ull, mykey = ~0ull;
    int outj = 0;
#pragma unroll 1
    for (int sp = 0; sp < KF / 4; sp++) {
        unsigned long long k0 = ~0ull, k1 = ~0ull, k2 = ~0ull, k3 = ~0ull;
        for (int i = lane; i < NN; i += 32) {
            float dx = xs[i] - qx, dy = ys[i] - qy, dz = zs[i] - qz;
            float d  = dx * dx + dy * dy + dz * dz;
            unsigned long long key =
                ((unsigned long long)__float_as_uint(d) << 32) | (unsigned)(i + 1);
            if (key > prev) ins4(key, k0, k1, k2, k3);
        }
#pragma unroll
        for (int t = 0; t < 4; t++) {
            unsigned long long best = wave_min_u64(k0);
            if (k0 == best) { k0 = k1; k1 = k2; k2 = k3; k3 = ~0ull; }
            prev = best;
            if (lane == outj + t) mykey = best;
        }
        outj += 4;
    }
    int idx = (int)(mykey & 0xffffffffull) - 1;
    float fx = xs[idx], fy = ys[idx], fz = zs[idx];
    long o = ((long)w * KF + lane) * 3;
    frames[o] = fx; frames[o + 1] = fy; frames[o + 2] = fz;
    xc[o] = fx - qx; xc[o + 1] = fy - qy; xc[o + 2] = fz - qz;
}

// ---------------- Kernel 3: KNN2 (4 nearest clean) -> grad_target ----------------
// One wave per frame point; 8 waves/block; 512 blocks per batch. Single scan:
// per-lane sorted top-4 provably contains the global top-4; 4 extraction rounds.
__global__ __launch_bounds__(256) void k_knn2(
    const float* __restrict__ clean, const float* __restrict__ frames,
    float* __restrict__ gt)
{
    __shared__ float xs[NN], ys[NN], zs[NN];   // 120 KB SoA cloud

    int tid = threadIdx.x;
    int b   = blockIdx.x >> 9;                 // 512 blocks per batch
    const float* src = clean + (long)b * NN * 3;
    for (int e = tid; e < NN; e += 256) {
        __builtin_prefetch(src + 3 * e + 768, 0, 0);
        xs[e] = src[3 * e];
        ys[e] = src[3 * e + 1];
        zs[e] = src[3 * e + 2];
    }
    __syncthreads();

    int wave = tid >> 5, lane = tid & 31;
    int w    = blockIdx.x * 8 + wave;          // frame-point index in [0, B*M*K)
    float fx = frames[w * 3], fy = frames[w * 3 + 1], fz = frames[w * 3 + 2];

    unsigned long long k0 = ~0ull, k1 = ~0ull, k2 = ~0ull, k3 = ~0ull;
    for (int i = lane; i < NN; i += 32) {
        float dx = xs[i] - fx, dy = ys[i] - fy, dz = zs[i] - fz;
        float d  = dx * dx + dy * dy + dz * dz;
        unsigned long long key =
            ((unsigned long long)__float_as_uint(d) << 32) | (unsigned)(i + 1);
        ins4(key, k0, k1, k2, k3);
    }

    float sx = 0.f, sy = 0.f, sz = 0.f;
#pragma unroll
    for (int t = 0; t < CC; t++) {
        unsigned long long best = wave_min_u64(k0);
        if (k0 == best) { k0 = k1; k1 = k2; k2 = k3; k3 = ~0ull; }
        int idx = (int)(best & 0xffffffffull) - 1;
        sx += xs[idx]; sy += ys[idx]; sz += zs[idx];
    }
    if (lane == 0) {
        const float inv = 1.0f / (float)CC;
        gt[w * 3 + 0] = sx * inv - fx;
        gt[w * 3 + 1] = sy * inv - fy;
        gt[w * 3 + 2] = sz * inv - fz;
    }
}

// ---------------- Kernel 4: score MLP (WMMA) + DSM loss ----------------
// 1024 row-tiles of 16; 8 waves/block, one tile per wave; grid 128 x 256.
__global__ __launch_bounds__(256) void k_score(
    const float* __restrict__ sW1, const float* __restrict__ sb1,
    const float* __restrict__ sW2, const float* __restrict__ sb2,
    const float* __restrict__ sW3, const float* __restrict__ sb3,
    const _Float16* __restrict__ featF16, const float* __restrict__ xcg,
    const float* __restrict__ gt, float* __restrict__ out)
{
    __shared__ __align__(16) _Float16 w1s[128 * 128];   // W1 rows 3..130 swizzled
    __shared__ __align__(16) _Float16 w2s[128 * 128];   // W2 swizzled
    __shared__ float w1c[3 * 128];
    __shared__ float b1s[128], b2s[128];
    __shared__ float w3s[128 * 3];
    __shared__ float b3s[3];
    __shared__ __align__(16) _Float16 hbuf[8][16 * 128]; // per-wave h1/h2 tile
    __shared__ __align__(16) _Float16 featrow[8][128];   // per-wave broadcast feature row
    __shared__ float xcs[8][16 * 3];

    int tid = threadIdx.x;
    load_swz(sW1, HD, 3, w1s, tid, 256);
    load_swz(sW2, HD, 0, w2s, tid, 256);
    for (int e = tid; e < 3 * 128; e += 256) w1c[e] = sW1[e];
    for (int e = tid; e < 128; e += 256) { b1s[e] = sb1[e]; b2s[e] = sb2[e]; }
    for (int e = tid; e < 128 * 3; e += 256) w3s[e] = sW3[e];
    if (tid < 3) b3s[tid] = sb3[tid];
    __syncthreads();

    int wave = tid >> 5, lane = tid & 31;
    int tile = blockIdx.x * 8 + wave;
    int r0   = tile * 16;
    int bm   = r0 / KF;   // constant within a 16-row tile (16 | 32)

    // stage broadcast feature row + local coords
    {
        const _Float16* src = featF16 + (long)bm * FD;
        for (int e = lane; e < 128; e += 32) featrow[wave][e] = src[e];
        for (int e = lane; e < 48;  e += 32) xcs[wave][e] = xcg[(long)r0 * 3 + e];
    }
    __syncthreads();

    int col_r = lane & 15;
    int rloc  = (lane >> 4) << 3;   // 0 or 8

    // ---- layer 1: bias + coord part (K=3, VALU) + feature part (WMMA) ----
    v8f acc[8];
#pragma unroll
    for (int c = 0; c < 8; c++) {
        int col = c * 16 + col_r;
        float bias = b1s[col];
#pragma unroll
        for (int v = 0; v < 8; v++) {
            int row = rloc + v;
            acc[c][v] = bias + xcs[wave][row * 3 + 0] * w1c[col]
                             + xcs[wave][row * 3 + 1] * w1c[128 + col]
                             + xcs[wave][row * 3 + 2] * w1c[256 + col];
        }
    }
#pragma unroll
    for (int kk = 0; kk < 4; kk++) {
        v16h a = frag_from_row128(featrow[wave], kk, lane);
#pragma unroll
        for (int c = 0; c < 8; c++) {
            v16h b = frag_b_swz(w1s, c, kk, lane);
            acc[c] = wmma16(a, b, acc[c]);
        }
    }
#pragma unroll
    for (int c = 0; c < 8; c++) {
        int col = c * 16 + col_r;
#pragma unroll
        for (int v = 0; v < 8; v++)
            hbuf[wave][(rloc + v) * 128 + col] = (_Float16)fmaxf(acc[c][v], 0.0f);
    }
    __syncthreads();

    // ---- layer 2: (16,128)@(128,128) WMMA ----
#pragma unroll
    for (int c = 0; c < 8; c++) {
        float bias = b2s[c * 16 + col_r];
#pragma unroll
        for (int v = 0; v < 8; v++) acc[c][v] = bias;
    }
#pragma unroll
    for (int kk = 0; kk < 4; kk++) {
        v16h a = frag_a_tile(hbuf[wave], kk, lane);
#pragma unroll
        for (int c = 0; c < 8; c++) {
            v16h b = frag_b_swz(w2s, c, kk, lane);
            acc[c] = wmma16(a, b, acc[c]);
        }
    }
    __syncthreads();
#pragma unroll
    for (int c = 0; c < 8; c++) {
        int col = c * 16 + col_r;
#pragma unroll
        for (int v = 0; v < 8; v++)
            hbuf[wave][(rloc + v) * 128 + col] = (_Float16)fmaxf(acc[c][v], 0.0f);
    }
    __syncthreads();

    // ---- layer 3 (128->3, VALU) + loss ----
    float part = 0.0f;
    if (lane < 16) {
        const _Float16* h = hbuf[wave] + lane * 128;
        float g0 = b3s[0], g1 = b3s[1], g2 = b3s[2];
        for (int k2 = 0; k2 < 128; k2++) {
            float hv = (float)h[k2];
            g0 += hv * w3s[k2 * 3 + 0];
            g1 += hv * w3s[k2 * 3 + 1];
            g2 += hv * w3s[k2 * 3 + 2];
        }
        long r = r0 + lane;
        float d0 = gt[r * 3 + 0] - g0;
        float d1 = gt[r * 3 + 1] - g1;
        float d2 = gt[r * 3 + 2] - g2;
        part = (d0 * d0 + d1 * d1 + d2 * d2) * LOSS_SCALE;
    }
#pragma unroll
    for (int off = 16; off; off >>= 1) part += __shfl_xor(part, off, 32);
    if (lane == 0) atomicAdd(out, part);
}

// ---------------- launch ----------------

extern "C" void kernel_launch(void* const* d_in, const int* in_sizes, int n_in,
                              void* d_out, int out_size, void* d_ws, size_t ws_size,
                              hipStream_t stream)
{
    const float* noisy = (const float*)d_in[0];
    const float* clean = (const float*)d_in[1];
    const int*   pidx  = (const int*)d_in[2];
    const float* fW1 = (const float*)d_in[3];
    const float* fb1 = (const float*)d_in[4];
    const float* fW2 = (const float*)d_in[5];
    const float* fb2 = (const float*)d_in[6];
    const float* sW1 = (const float*)d_in[7];
    const float* sb1 = (const float*)d_in[8];
    const float* sW2 = (const float*)d_in[9];
    const float* sb2 = (const float*)d_in[10];
    const float* sW3 = (const float*)d_in[11];
    const float* sb3 = (const float*)d_in[12];

    char* w = (char*)d_ws;
    float*    qws     = (float*)(w + 0);           //   6 KB  (512*3 f32)
    _Float16* featF16 = (_Float16*)(w + 8192);     // 128 KB  (512*128 f16)
    float*    frames  = (float*)(w + 139264);      // 192 KB  (16384*3 f32)
    float*    xc      = (float*)(w + 335872);      // 192 KB
    float*    gt      = (float*)(w + 532480);      // 192 KB
    float*    out     = (float*)d_out;

    hipMemsetAsync(out, 0, sizeof(float), stream);

    k_feat<<<dim3(32),   dim3(256), 0, stream>>>(noisy, pidx, fW1, fb1, fW2, fb2, qws, featF16);
    k_knn1<<<dim3(64),   dim3(256), 0, stream>>>(noisy, qws, frames, xc);
    k_knn2<<<dim3(2048), dim3(256), 0, stream>>>(clean, frames, gt);
    k_score<<<dim3(128), dim3(256), 0, stream>>>(sW1, sb1, sW2, sb2, sW3, sb3,
                                                 featF16, xc, gt, out);
}
